// FluxSingleTransformerBlock_89953795048084
// MI455X (gfx1250) — matrix operand order repaired and checked by reading, not compiled, not run on hardware
//
#include <hip/hip_runtime.h>
#include <math.h>

#define DIMD   3072
#define HEADS  24
#define HD     128
#define SEQ    4608
#define D3     (3 * DIMD)
#define D4     (4 * DIMD)
#define D5     (5 * DIMD)
#define LN_EPS  1e-6f
#define RMS_EPS 1e-5f

typedef __attribute__((ext_vector_type(16))) __bf16 v16bf;
typedef __attribute__((ext_vector_type(8)))  __bf16 bf16x8;
typedef __attribute__((ext_vector_type(8)))  float  v8f;

// ---------- bf16 helpers (round-to-nearest-even) ----------
__device__ __forceinline__ unsigned short f2bf_u(float f) {
  union { float f; unsigned u; } v; v.f = f;
  unsigned u = v.u;
  u += 0x7FFFu + ((u >> 16) & 1u);
  return (unsigned short)(u >> 16);
}
__device__ __forceinline__ __bf16 f2bf(float f) {
  unsigned short u = f2bf_u(f);
  __bf16 r; __builtin_memcpy(&r, &u, 2);
  return r;
}

// ---------- WMMA fragment loaders (contiguous k order, shared by A and B) ----------
// A fragment 16x32 bf16: lane row = lane%16; per half (lane/16):
//   elements i -> k = k0 + half*16 + i (contiguous 16, matches B)
__device__ __forceinline__ v16bf load_a_frag(const unsigned short* rowk0, int half) {
  bf16x8 lo = *(const bf16x8*)(rowk0 + half * 16);
  bf16x8 hi = *(const bf16x8*)(rowk0 + half * 16 + 8);
  v16bf r;
#pragma unroll
  for (int i = 0; i < 8; ++i) { r[i] = lo[i]; r[i + 8] = hi[i]; }
  return r;
}
// B fragment 32x16 bf16 (B[k][n] = W[n][k]): lane col = lane%16,
// elements i -> k = k0 + (lane/16)*16 + i  (one contiguous 32B load)
__device__ __forceinline__ v16bf load_b_frag(const unsigned short* rowk0, int half) {
  return *(const v16bf*)(rowk0 + half * 16);
}

// =====================================================================
// 1) AdaLN modulation GEMV: emb[n] = sum_k silu(te[k]) * w[n,k] + b[n]
// =====================================================================
__global__ __launch_bounds__(256) void adaln_kernel(
    const float* __restrict__ te, const float* __restrict__ w,
    const float* __restrict__ b, float* __restrict__ emb) {
  const int n = blockIdx.x;
  const int t = threadIdx.x;
  const float* wrow = w + (size_t)n * DIMD;
  float acc = 0.f;
#pragma unroll
  for (int i = 0; i < 12; ++i) {
    int k = t + i * 256;
    float x = te[k];
    float s = x / (1.f + __expf(-x));
    acc += s * wrow[k];
  }
#pragma unroll
  for (int o = 16; o > 0; o >>= 1) acc += __shfl_xor(acc, o);
  __shared__ float sh[8];
  if ((t & 31) == 0) sh[t >> 5] = acc;
  __syncthreads();
  if (t == 0) {
    float s = 0.f;
#pragma unroll
    for (int i = 0; i < 8; ++i) s += sh[i];
    emb[n] = s + b[n];
  }
}

// =====================================================================
// 2) LayerNorm + (1+scale)*x + shift  ->  bf16
// =====================================================================
__global__ __launch_bounds__(256) void ln_mod_kernel(
    const float* __restrict__ comb, const float* __restrict__ emb,
    unsigned short* __restrict__ xh) {
  const int row = blockIdx.x;
  const int t = threadIdx.x;
  const float* crow = comb + (size_t)row * DIMD;
  float vals[12];
  float s = 0.f, ss = 0.f;
#pragma unroll
  for (int i = 0; i < 12; ++i) {
    float x = crow[t + i * 256];
    vals[i] = x; s += x; ss += x * x;
  }
#pragma unroll
  for (int o = 16; o > 0; o >>= 1) { s += __shfl_xor(s, o); ss += __shfl_xor(ss, o); }
  __shared__ float sh[16];
  if ((t & 31) == 0) { sh[t >> 5] = s; sh[8 + (t >> 5)] = ss; }
  __syncthreads();
  float ts = 0.f, tss = 0.f;
#pragma unroll
  for (int i = 0; i < 8; ++i) { ts += sh[i]; tss += sh[8 + i]; }
  const float mu = ts / (float)DIMD;
  const float var = tss / (float)DIMD - mu * mu;
  const float inv = rsqrtf(var + LN_EPS);
  const float* shift = emb;
  const float* scale = emb + DIMD;
  unsigned short* xrow = xh + (size_t)row * DIMD;
#pragma unroll
  for (int i = 0; i < 12; ++i) {
    int n = t + i * 256;
    float x = (vals[i] - mu) * inv * (1.f + scale[n]) + shift[n];
    xrow[n] = f2bf_u(x);
  }
}

// =====================================================================
// 3) f32 -> bf16 weight conversion (grid-stride)
// =====================================================================
__global__ __launch_bounds__(256) void cvt_kernel(
    const float* __restrict__ in, unsigned short* __restrict__ out, long long n) {
  long long i = (long long)blockIdx.x * blockDim.x + threadIdx.x;
  long long stride = (long long)gridDim.x * blockDim.x;
  for (; i < n; i += stride) out[i] = f2bf_u(in[i]);
}

// =====================================================================
// 4) WMMA GEMM: C[M,N] = A[M,K](bf16) * W[N,K]^T(bf16), f32 accum.
//    Block tile = 128x64, 4 waves; each wave: 32 rows x 64 cols
//    (2 A-frags reuse 4 B-frags -> 8 WMMAs / k-step).
//    Fragments are double-buffered so WMMAs consume loads issued one
//    iteration earlier (partial s_wait instead of full barrier).
//    mode 0: Cf[m*ldc+n] = acc
//    mode 1: Cb[m*ldc+n] = bf16(gelu_tanh(acc + bias[n]))
//    mode 2: Cf[m*ldc+n] = resid[m*ldc+n] + gate[n]*(acc + bias[n])
// =====================================================================
__global__ __launch_bounds__(128) void gemm_bf16_kernel(
    const unsigned short* __restrict__ A, const unsigned short* __restrict__ W,
    int M, int N, int K, int mode,
    float* __restrict__ Cf, unsigned short* __restrict__ Cb, int ldc,
    const float* __restrict__ bias, const float* __restrict__ gate,
    const float* __restrict__ resid) {
  const int wave = threadIdx.x >> 5;
  const int lane = threadIdx.x & 31;
  const int hf = lane >> 4;
  const int ln = lane & 15;
  const int m0 = blockIdx.y * 128 + wave * 32;
  const int n0 = blockIdx.x * 64;

  const unsigned short* arow0 = A + (size_t)(m0 + ln) * K;
  const unsigned short* arow1 = A + (size_t)(m0 + 16 + ln) * K;
  const unsigned short* wrow0 = W + (size_t)(n0 + ln) * K;
  const unsigned short* wrow1 = W + (size_t)(n0 + 16 + ln) * K;
  const unsigned short* wrow2 = W + (size_t)(n0 + 32 + ln) * K;
  const unsigned short* wrow3 = W + (size_t)(n0 + 48 + ln) * K;

  v8f acc[2][4];
#pragma unroll
  for (int u = 0; u < 2; ++u)
#pragma unroll
    for (int t = 0; t < 4; ++t)
#pragma unroll
      for (int r = 0; r < 8; ++r) acc[u][t][r] = 0.f;

  // prologue: load k0 = 0 fragments
  v16bf a0 = load_a_frag(arow0, hf);
  v16bf a1 = load_a_frag(arow1, hf);
  v16bf b0 = load_b_frag(wrow0, hf);
  v16bf b1 = load_b_frag(wrow1, hf);
  v16bf b2 = load_b_frag(wrow2, hf);
  v16bf b3 = load_b_frag(wrow3, hf);

  for (int k0 = 0; k0 < K; k0 += 32) {
    const int kn = (k0 + 32 < K) ? (k0 + 32) : 0;  // harmless dummy reload on last iter
    __builtin_prefetch(arow0 + k0 + 256, 0, 0);
    v16bf a0n = load_a_frag(arow0 + kn, hf);
    v16bf a1n = load_a_frag(arow1 + kn, hf);
    v16bf b0n = load_b_frag(wrow0 + kn, hf);
    v16bf b1n = load_b_frag(wrow1 + kn, hf);
    v16bf b2n = load_b_frag(wrow2 + kn, hf);
    v16bf b3n = load_b_frag(wrow3 + kn, hf);

    acc[0][0] = __builtin_amdgcn_wmma_f32_16x16x32_bf16(false, a0, false, b0, (short)0, acc[0][0], false, false);
    acc[0][1] = __builtin_amdgcn_wmma_f32_16x16x32_bf16(false, a0, false, b1, (short)0, acc[0][1], false, false);
    acc[0][2] = __builtin_amdgcn_wmma_f32_16x16x32_bf16(false, a0, false, b2, (short)0, acc[0][2], false, false);
    acc[0][3] = __builtin_amdgcn_wmma_f32_16x16x32_bf16(false, a0, false, b3, (short)0, acc[0][3], false, false);
    acc[1][0] = __builtin_amdgcn_wmma_f32_16x16x32_bf16(false, a1, false, b0, (short)0, acc[1][0], false, false);
    acc[1][1] = __builtin_amdgcn_wmma_f32_16x16x32_bf16(false, a1, false, b1, (short)0, acc[1][1], false, false);
    acc[1][2] = __builtin_amdgcn_wmma_f32_16x16x32_bf16(false, a1, false, b2, (short)0, acc[1][2], false, false);
    acc[1][3] = __builtin_amdgcn_wmma_f32_16x16x32_bf16(false, a1, false, b3, (short)0, acc[1][3], false, false);

    a0 = a0n; a1 = a1n; b0 = b0n; b1 = b1n; b2 = b2n; b3 = b3n;
  }

#pragma unroll
  for (int u = 0; u < 2; ++u) {
#pragma unroll
    for (int t = 0; t < 4; ++t) {
      const int n = n0 + t * 16 + ln;
      if (mode == 0) {
#pragma unroll
        for (int r = 0; r < 8; ++r) {
          const int m = m0 + u * 16 + hf * 8 + r;
          Cf[(size_t)m * ldc + n] = acc[u][t][r];
        }
      } else if (mode == 1) {
        const float bn = bias[n];
#pragma unroll
        for (int r = 0; r < 8; ++r) {
          const int m = m0 + u * 16 + hf * 8 + r;
          float x = acc[u][t][r] + bn;
          // tanh-approx gelu via exp: tanh(u) = 1 - 2/(exp(2u)+1)
          float uarg = 0.7978845608028654f * (x + 0.044715f * x * x * x);
          float th = 1.f - 2.f / (__expf(2.f * uarg) + 1.f);
          float g = 0.5f * x * (1.f + th);
          Cb[(size_t)m * ldc + n] = f2bf_u(g);
        }
      } else {
        const float bn = bias[n];
        const float gn = gate[n];
#pragma unroll
        for (int r = 0; r < 8; ++r) {
          const int m = m0 + u * 16 + hf * 8 + r;
          const size_t idx = (size_t)m * ldc + n;
          Cf[idx] = resid[idx] + gn * (acc[u][t][r] + bn);
        }
      }
    }
  }
}

// =====================================================================
// 5) RMSNorm + RoPE (+ 1/sqrt(HD) folded into q) -> bf16 q,k; V^T -> bf16
// =====================================================================
__global__ __launch_bounds__(128) void rope_kernel(
    const float* __restrict__ qf, const float* __restrict__ kf,
    const float* __restrict__ vf, const float* __restrict__ cosT,
    const float* __restrict__ sinT, const float* __restrict__ qw,
    const float* __restrict__ kw, unsigned short* __restrict__ qh,
    unsigned short* __restrict__ kh, unsigned short* __restrict__ vt) {
  const int row = blockIdx.x, h = blockIdx.y, d = threadIdx.x;
  const size_t idx = (size_t)row * DIMD + (size_t)h * HD + d;
  float q = qf[idx], k = kf[idx], v = vf[idx];
  float sq = q * q, sk = k * k;
#pragma unroll
  for (int o = 16; o > 0; o >>= 1) { sq += __shfl_xor(sq, o); sk += __shfl_xor(sk, o); }
  __shared__ float sh[8];
  if ((d & 31) == 0) { sh[d >> 5] = sq; sh[4 + (d >> 5)] = sk; }
  __syncthreads();
  const float tq = sh[0] + sh[1] + sh[2] + sh[3];
  const float tk = sh[4] + sh[5] + sh[6] + sh[7];
  const float qn = q * rsqrtf(tq / (float)HD + RMS_EPS) * qw[d];
  const float kn = k * rsqrtf(tk / (float)HD + RMS_EPS) * kw[d];
  __shared__ float bq[HD], bk[HD];
  bq[d] = qn; bk[d] = kn;
  __syncthreads();
  const float rq = (d & 1) ? bq[d - 1] : -bq[d + 1];
  const float rk = (d & 1) ? bk[d - 1] : -bk[d + 1];
  const float c = cosT[row * HD + d], s = sinT[row * HD + d];
  const float qr = (qn * c + rq * s) * 0.08838834764831845f; // 1/sqrt(128)
  const float kr = kn * c + rk * s;
  qh[idx] = f2bf_u(qr);
  kh[idx] = f2bf_u(kr);
  vt[(size_t)(h * HD + d) * SEQ + row] = f2bf_u(v);
}

// =====================================================================
// 6) Flash attention (per head, 64-query block, 4 waves x 16 queries).
//    Scores computed transposed: ST(16key x 16q) = Kblk(16x32) x Q^T(32x16)
//    PV: out^T(16d x 16q) = V^T(16x32keys) x P^T(32x16)
// =====================================================================
__global__ __launch_bounds__(128) void attn_kernel(
    const unsigned short* __restrict__ qh, const unsigned short* __restrict__ kh,
    const unsigned short* __restrict__ vt, unsigned short* __restrict__ addb) {
  const int h = blockIdx.y;
  const int wave = threadIdx.x >> 5;
  const int lane = threadIdx.x & 31;
  const int hf = lane >> 4;
  const int ln = lane & 15;
  const int mrow = blockIdx.x * 64 + wave * 16 + ln; // this lane's query row

  // Q^T B-fragments (4 k-steps over head dim)
  const unsigned short* qrow = qh + (size_t)mrow * DIMD + (size_t)h * HD;
  v16bf qb[4];
#pragma unroll
  for (int ds = 0; ds < 4; ++ds) qb[ds] = load_b_frag(qrow + ds * 32, hf);

  v8f oacc[8];
#pragma unroll
  for (int t = 0; t < 8; ++t)
#pragma unroll
    for (int r = 0; r < 8; ++r) oacc[t][r] = 0.f;
  float run_max = -3.0e38f, run_sum = 0.f;

  for (int kb = 0; kb < SEQ; kb += 32) {
    // scores^T: two 16-key tiles
    v8f st[2];
#pragma unroll
    for (int t = 0; t < 2; ++t) {
#pragma unroll
      for (int r = 0; r < 8; ++r) st[t][r] = 0.f;
      const unsigned short* krow =
          kh + (size_t)(kb + t * 16 + ln) * DIMD + (size_t)h * HD;
#pragma unroll
      for (int ds = 0; ds < 4; ++ds) {
        v16bf ka = load_a_frag(krow + ds * 32, hf);
        st[t] = __builtin_amdgcn_wmma_f32_16x16x32_bf16(false, ka, false, qb[ds],
                                                        (short)0, st[t], false, false);
      }
    }
    // online softmax over 32 keys for this lane's query column
    float lm = -3.0e38f;
#pragma unroll
    for (int t = 0; t < 2; ++t)
#pragma unroll
      for (int r = 0; r < 8; ++r) lm = fmaxf(lm, st[t][r]);
    lm = fmaxf(lm, __shfl_xor(lm, 16));
    const float nm = fmaxf(run_max, lm);
    const float alpha = __expf(run_max - nm);
    run_max = nm;
    float p0[8], p1[8], psum = 0.f;
#pragma unroll
    for (int r = 0; r < 8; ++r) {
      p0[r] = __expf(st[0][r] - nm);
      p1[r] = __expf(st[1][r] - nm);
      psum += p0[r] + p1[r];
    }
    psum += __shfl_xor(psum, 16);
    run_sum = run_sum * alpha + psum;
#pragma unroll
    for (int t = 0; t < 8; ++t)
#pragma unroll
      for (int r = 0; r < 8; ++r) oacc[t][r] *= alpha;
    // assemble P^T B-fragment (exchange halves via xor-16)
    v16bf pb;
#pragma unroll
    for (int r = 0; r < 8; ++r) {
      const float o0 = __shfl_xor(p0[r], 16);
      const float o1 = __shfl_xor(p1[r], 16);
      const float lo = hf ? o1 : p0[r];
      const float hi = hf ? p1[r] : o0;
      pb[r] = f2bf(lo);
      pb[r + 8] = f2bf(hi);
    }
    // out^T += V^T x P^T over 8 d-tiles (V^T rows are contiguous over keys)
#pragma unroll
    for (int t = 0; t < 8; ++t) {
      const unsigned short* vrow =
          vt + (size_t)(h * HD + t * 16 + ln) * SEQ + kb;
      v16bf va = load_a_frag(vrow, hf);
      oacc[t] = __builtin_amdgcn_wmma_f32_16x16x32_bf16(false, va, false, pb,
                                                        (short)0, oacc[t], false, false);
    }
  }

  const float inv = 1.f / run_sum;
  unsigned short* orow = addb + (size_t)mrow * D5 + (size_t)h * HD;
#pragma unroll
  for (int t = 0; t < 8; ++t)
#pragma unroll
    for (int r = 0; r < 8; ++r)
      orow[t * 16 + hf * 8 + r] = f2bf_u(oacc[t][r] * inv);
}

// =====================================================================
// Host orchestration
// =====================================================================
extern "C" void kernel_launch(void* const* d_in, const int* in_sizes, int n_in,
                              void* d_out, int out_size, void* d_ws, size_t ws_size,
                              hipStream_t stream) {
  (void)in_sizes; (void)n_in; (void)out_size; (void)ws_size;
  const float* combined  = (const float*)d_in[0];
  const float* time_emb  = (const float*)d_in[1];
  const float* rope_cos  = (const float*)d_in[2];
  const float* rope_sin  = (const float*)d_in[3];
  const float* adaln_w   = (const float*)d_in[4];
  const float* adaln_b   = (const float*)d_in[5];
  const float* mlp_w     = (const float*)d_in[6];
  const float* mlp_b     = (const float*)d_in[7];
  const float* wq        = (const float*)d_in[8];
  const float* wk        = (const float*)d_in[9];
  const float* wv        = (const float*)d_in[10];
  const float* qnw       = (const float*)d_in[11];
  const float* knw       = (const float*)d_in[12];
  const float* out_w     = (const float*)d_in[13];
  const float* out_b     = (const float*)d_in[14];
  float* out = (float*)d_out;

  char* ws = (char*)d_ws;
  size_t off = 0;
  auto alloc = [&](size_t bytes) -> void* {
    void* p = ws + off;
    off += (bytes + 255) & ~(size_t)255;
    return p;
  };
  float*          emb   = (float*)alloc((size_t)D3 * 4);
  unsigned short* xh    = (unsigned short*)alloc((size_t)SEQ * DIMD * 2);
  unsigned short* wqh   = (unsigned short*)alloc((size_t)DIMD * DIMD * 2);
  unsigned short* wkh   = (unsigned short*)alloc((size_t)DIMD * DIMD * 2);
  unsigned short* wvh   = (unsigned short*)alloc((size_t)DIMD * DIMD * 2);
  unsigned short* mlpwh = (unsigned short*)alloc((size_t)D4 * DIMD * 2);
  unsigned short* outwh = (unsigned short*)alloc((size_t)DIMD * D5 * 2);
  float*          qf    = (float*)alloc((size_t)SEQ * DIMD * 4);
  float*          kf    = (float*)alloc((size_t)SEQ * DIMD * 4);
  float*          vf    = (float*)alloc((size_t)SEQ * DIMD * 4);
  unsigned short* qhb   = (unsigned short*)alloc((size_t)SEQ * DIMD * 2);
  unsigned short* khb   = (unsigned short*)alloc((size_t)SEQ * DIMD * 2);
  unsigned short* vtb   = (unsigned short*)alloc((size_t)SEQ * DIMD * 2);
  // additional[S, 5D] bf16 aliases the dead qf/kf/vf region (141.6MB <= 169.9MB);
  // all reads of qf/kf/vf (rope_kernel) complete before anything writes `add`.
  unsigned short* add = (unsigned short*)qf;

  // 1) AdaLN modulation
  adaln_kernel<<<D3, 256, 0, stream>>>(time_emb, adaln_w, adaln_b, emb);
  // 2) LayerNorm + modulate -> bf16 x
  ln_mod_kernel<<<SEQ, 256, 0, stream>>>(combined, emb, xh);
  // 3) Weight conversions to bf16
  cvt_kernel<<<4096, 256, 0, stream>>>(wq,    wqh,   (long long)DIMD * DIMD);
  cvt_kernel<<<4096, 256, 0, stream>>>(wk,    wkh,   (long long)DIMD * DIMD);
  cvt_kernel<<<4096, 256, 0, stream>>>(wv,    wvh,   (long long)DIMD * DIMD);
  cvt_kernel<<<4096, 256, 0, stream>>>(mlp_w, mlpwh, (long long)D4 * DIMD);
  cvt_kernel<<<4096, 256, 0, stream>>>(out_w, outwh, (long long)DIMD * D5);
  // 4) QKV projections (WMMA, f32 out)
  dim3 gqkv(DIMD / 64, SEQ / 128);
  gemm_bf16_kernel<<<gqkv, 128, 0, stream>>>(xh, wqh, SEQ, DIMD, DIMD, 0,
                                             qf, nullptr, DIMD, nullptr, nullptr, nullptr);
  gemm_bf16_kernel<<<gqkv, 128, 0, stream>>>(xh, wkh, SEQ, DIMD, DIMD, 0,
                                             kf, nullptr, DIMD, nullptr, nullptr, nullptr);
  gemm_bf16_kernel<<<gqkv, 128, 0, stream>>>(xh, wvh, SEQ, DIMD, DIMD, 0,
                                             vf, nullptr, DIMD, nullptr, nullptr, nullptr);
  // 5) RMSNorm + RoPE -> bf16 q,k ; V^T -> bf16
  rope_kernel<<<dim3(SEQ, HEADS), 128, 0, stream>>>(qf, kf, vf, rope_cos, rope_sin,
                                                    qnw, knw, qhb, khb, vtb);
  // 6) MLP up-proj + gelu -> additional[:, 3072:15360]
  dim3 gmlp(D4 / 64, SEQ / 128);
  gemm_bf16_kernel<<<gmlp, 128, 0, stream>>>(xh, mlpwh, SEQ, D4, DIMD, 1,
                                             nullptr, add + DIMD, D5, mlp_b, nullptr, nullptr);
  // 7) Attention -> additional[:, 0:3072]
  attn_kernel<<<dim3(SEQ / 64, HEADS), 128, 0, stream>>>(qhb, khb, vtb, add);
  // 8) Output projection + gated residual -> out
  dim3 gout(DIMD / 64, SEQ / 128);
  gemm_bf16_kernel<<<gout, 128, 0, stream>>>(add, outwh, SEQ, DIMD, D5, 2,
                                             out, nullptr, DIMD, out_b, emb + 2 * DIMD, combined);
}
